// ScaledDotProductAttention_36584531427831
// MI455X (gfx1250) — compile-verified, matrix operand order checked by reading, not run on hardware
//
#include <hip/hip_runtime.h>

typedef __attribute__((ext_vector_type(16))) _Float16 v16h;
typedef __attribute__((ext_vector_type(8)))  _Float16 v8h;
typedef __attribute__((ext_vector_type(8)))  float    v8f;

#define ATT_B 2
#define ATT_H 16
#define ATT_S 2048
#define ATT_D 64
#define BH    (ATT_B * ATT_H)
#define NELEM ((size_t)BH * ATT_S * ATT_D)   // 4,194,304 per tensor

// exp(q.k / sqrt(64)) == exp2( (q * log2(e)/8) . k )  -> baked into Qh
#define EXP2_SCALE 0.18033688011112042f

#if __has_builtin(__builtin_amdgcn_exp2f)
__device__ __forceinline__ float fast_exp2(float x) { return __builtin_amdgcn_exp2f(x); }
#else
__device__ __forceinline__ float fast_exp2(float x) {
  float r; asm volatile("v_exp_f32 %0, %1" : "=v"(r) : "v"(x)); return r;
}
#endif

#if __has_builtin(__builtin_amdgcn_rcpf)
__device__ __forceinline__ float fast_rcp(float x) { return __builtin_amdgcn_rcpf(x); }
#else
__device__ __forceinline__ float fast_rcp(float x) {
  float r; asm volatile("v_rcp_f32 %0, %1" : "=v"(r) : "v"(x)); return r;
}
#endif

__device__ __forceinline__ v16h cat8(v8h a, v8h b) {
  return __builtin_shufflevector(a, b, 0,1,2,3,4,5,6,7,8,9,10,11,12,13,14,15);
}

__device__ __forceinline__ void lds_phase_fence() {
  __builtin_amdgcn_wave_barrier();
  asm volatile("s_wait_dscnt 0x0" ::: "memory");
  __builtin_amdgcn_wave_barrier();
}

// -------- pre-pass 1: fp32 -> fp16 elementwise with optional scaling --------
__global__ __launch_bounds__(256)
void cvt_f16_kernel(const float* __restrict__ src, _Float16* __restrict__ dst,
                    float mul) {
  size_t i = (size_t)blockIdx.x * 256 + threadIdx.x;   // one v8h per thread
  const float4* s4 = (const float4*)src;
  float4 a = s4[i * 2 + 0];
  float4 b = s4[i * 2 + 1];
  v8h o;
  o[0] = (_Float16)(a.x * mul); o[1] = (_Float16)(a.y * mul);
  o[2] = (_Float16)(a.z * mul); o[3] = (_Float16)(a.w * mul);
  o[4] = (_Float16)(b.x * mul); o[5] = (_Float16)(b.y * mul);
  o[6] = (_Float16)(b.z * mul); o[7] = (_Float16)(b.w * mul);
  *(v8h*)(dst + i * 8) = o;
}

// -------- pre-pass 2: V [bh][s][d] fp32 -> Vt [bh][d][s] fp16 ---------------
__global__ __launch_bounds__(256)
void vtrans_f16_kernel(const float* __restrict__ V, _Float16* __restrict__ Vt) {
  constexpr int S = ATT_S, D = ATT_D;
  __shared__ __align__(16) _Float16 t[64][72];
  const int bh = blockIdx.x >> 5;
  const int s0 = (blockIdx.x & 31) << 6;
  const int tid = threadIdx.x;
#pragma unroll
  for (int i = 0; i < 4; ++i) {
    int idx = i * 256 + tid;
    int r  = idx >> 4;
    int c4 = (idx & 15) << 2;
    float4 v4 = *(const float4*)(V + ((size_t)bh * S + s0 + r) * D + c4);
    t[c4 + 0][r] = (_Float16)v4.x;
    t[c4 + 1][r] = (_Float16)v4.y;
    t[c4 + 2][r] = (_Float16)v4.z;
    t[c4 + 3][r] = (_Float16)v4.w;
  }
  __syncthreads();
#pragma unroll
  for (int i = 0; i < 2; ++i) {
    int ch = i * 256 + tid;
    int d  = ch >> 3;
    int s8 = (ch & 7) << 3;
    *(v8h*)(Vt + ((size_t)bh * D + d) * S + s0 + s8) = *(const v8h*)&t[d][s8];
  }
}

// ---- one 32-wide k-tile: S^T = K.Q^T (C-layout of S^T == A-layout of P) ----
template<bool MASKED>
__device__ __forceinline__ void tile_step(int k0, int q16, int ln, int hi,
                                          const _Float16* kp, const _Float16* vp,
                                          float* ap,
                                          const v16h& qb0, const v16h& qb1,
                                          v8f (&acc)[4], float& l) {
  constexpr int S = ATT_S, D = ATT_D;
  v8f ct[2];
#pragma unroll
  for (int t = 0; t < 2; ++t) {
    const _Float16* kr = kp + t * (16 * D);
    v16h ka0 = cat8(*(const v8h*)(kr),      *(const v8h*)(kr + 16));
    v16h ka1 = cat8(*(const v8h*)(kr + 32), *(const v8h*)(kr + 48));
    v8f cc = {};
    cc = __builtin_amdgcn_wmma_f32_16x16x32_f16(false, ka0, false, qb0, (short)0, cc, false, false);
    cc = __builtin_amdgcn_wmma_f32_16x16x32_f16(false, ka1, false, qb1, (short)0, cc, false, false);
    ct[t] = cc;
  }

  float p[2][8];
  v16h pa;   // A-fragment of P for the PV WMMA, directly from registers
#pragma unroll
  for (int t = 0; t < 2; ++t) {
#pragma unroll
    for (int g = 0; g < 8; ++g) {
      float v = fast_exp2(ct[t][g]);        // scale pre-baked into Qh
      if (MASKED) {
        int kg = k0 + t * 16 + hi * 8 + g;
        v = (kg <= q16 + ln) ? v : 0.0f;    // masked -> 0 (== softmax of -1e9)
      }
      p[t][g] = v;
      l += v;
      pa[t * 8 + g] = (_Float16)v;
    }
  }

  // unnormalized attn: each lane stores 2x16B contiguous runs per sub-tile
#pragma unroll
  for (int t = 0; t < 2; ++t) {
    float4 f0 = {p[t][0], p[t][1], p[t][2], p[t][3]};
    float4 f1 = {p[t][4], p[t][5], p[t][6], p[t][7]};
    *(float4*)(ap + t * 16)     = f0;
    *(float4*)(ap + t * 16 + 4) = f1;
  }

  // PV accumulate: B = Vt straight from global (fp16, contiguous along k)
#pragma unroll
  for (int nb = 0; nb < 4; ++nb) {
    const _Float16* vr = vp + nb * (16 * S);
    v16h vb = cat8(*(const v8h*)(vr), *(const v8h*)(vr + 16));
    acc[nb] = __builtin_amdgcn_wmma_f32_16x16x32_f16(false, pa, false, vb, (short)0, acc[nb], false, false);
  }
}

// ---------------- main: causal SDPA, one wave = 16 q rows -------------------
__global__ __launch_bounds__(128)
void sdpa_causal_wmma_kernel(const _Float16* __restrict__ Qh,
                             const _Float16* __restrict__ Kh,
                             const _Float16* __restrict__ Vt,
                             float* __restrict__ ctx,
                             float* __restrict__ attn) {
  constexpr int S = ATT_S, D = ATT_D;

  __shared__ float ROWSUM[4][16];

  const int tid  = threadIdx.x;
  const int lane = tid & 31;
  const int w    = tid >> 5;
  const int ln   = lane & 15;
  const int hi   = lane >> 4;

  const int bh  = blockIdx.x >> 5;
  const int qb  = (blockIdx.x & 31) << 6;
  const int q16 = qb + w * 16;
  const int kunm = (q16 >> 5) << 5;        // tiles below kunm are mask-free
  const int kend = kunm + 32;              // exactly one diagonal (masked) tile

  // Q fragment doubles as the WMMA B operand of S^T = K.Q^T
  const _Float16* qr = Qh + ((size_t)bh * S + q16 + ln) * D + hi * 8;
  const v16h qb0 = cat8(*(const v8h*)(qr),      *(const v8h*)(qr + 16));
  const v16h qb1 = cat8(*(const v8h*)(qr + 32), *(const v8h*)(qr + 48));

  // marching pointers; all other offsets are compile-time immediates
  const _Float16* kp = Kh + ((size_t)bh * S) * D + (size_t)ln * D + hi * 8;
  const _Float16* vp = Vt + ((size_t)bh * D) * S + (size_t)ln * S + hi * 8;
  float*          ap = attn + ((size_t)(bh * S + q16 + ln)) * S + hi * 8;

  v8f acc[4] = {};
  float l = 0.f;

  for (int k0 = 0; k0 < kunm; k0 += 32) {
    tile_step<false>(k0, q16, ln, hi, kp, vp, ap, qb0, qb1, acc, l);
    kp += 32 * D;
    vp += 32;
    ap += 32;
  }
  tile_step<true>(kunm, q16, ln, hi, kp, vp, ap, qb0, qb1, acc, l);

  // ---- causal upper triangle: zero-fill (output buffer is poisoned) ----
  {
    float4 z = {0.f, 0.f, 0.f, 0.f};
    float* zp = attn + ((size_t)(bh * S + q16 + ln)) * S + hi * 16 + kend;
    for (int k0 = kend; k0 < S; k0 += 32) {
      *(float4*)(zp)      = z;
      *(float4*)(zp + 4)  = z;
      *(float4*)(zp + 8)  = z;
      *(float4*)(zp + 12) = z;
      zp += 32;
    }
  }

  // ---- row sum: combine the two hi-halves of each q row ----
  {
    float v = l + __shfl_xor(l, 16);
    if (hi == 0) ROWSUM[w][ln] = v;
  }
  lds_phase_fence();

  // ---- context = acc / l (acc is in normal ctx orientation) ----
  float rinv[8];
#pragma unroll
  for (int g = 0; g < 8; ++g) rinv[g] = fast_rcp(ROWSUM[w][g + hi * 8]);
#pragma unroll
  for (int nb = 0; nb < 4; ++nb) {
#pragma unroll
    for (int g = 0; g < 8; ++g) {
      int qg = q16 + g + hi * 8;
      ctx[((size_t)(bh * S + qg)) * D + nb * 16 + ln] = acc[nb][g] * rinv[g];
    }
  }

  // ---- normalize this wave's attn rows in place (L2-hot) ----
  for (int r = 0; r < 16; ++r) {
    int qg = q16 + r;
    float rv = fast_rcp(ROWSUM[w][r]);
    float* row = attn + ((size_t)(bh * S + qg)) * S;
    int nch = (qg + 128) >> 7;               // ceil((qg+1)/128); tail is zeros
    for (int ch = 0; ch < nch; ++ch) {
      float4* p4 = (float4*)(row + ch * 128 + lane * 4);
      float4 v = *p4;
      v.x *= rv; v.y *= rv; v.z *= rv; v.w *= rv;
      *p4 = v;
    }
  }
}

extern "C" void kernel_launch(void* const* d_in, const int* in_sizes, int n_in,
                              void* d_out, int out_size, void* d_ws, size_t ws_size,
                              hipStream_t stream) {
  (void)in_sizes; (void)n_in; (void)out_size; (void)ws_size;
  const float* Q = (const float*)d_in[0];
  const float* K = (const float*)d_in[1];
  const float* V = (const float*)d_in[2];
  float* out  = (float*)d_out;
  float* ctx  = out;                         // [B,H,S,D]
  float* attn = out + NELEM;                 // [B,H,S,S]

  _Float16* Qh = (_Float16*)d_ws;            // 8 MB, pre-scaled by log2(e)/8
  _Float16* Kh = Qh + NELEM;                 // 8 MB
  _Float16* Vt = Kh + NELEM;                 // 8 MB, transposed [bh][d][s]

  const int cvtBlocks = (int)(NELEM / (256 * 8));
  hipLaunchKernelGGL(cvt_f16_kernel,    dim3(cvtBlocks), dim3(256), 0, stream, Q, Qh, EXP2_SCALE);
  hipLaunchKernelGGL(cvt_f16_kernel,    dim3(cvtBlocks), dim3(256), 0, stream, K, Kh, 1.0f);
  hipLaunchKernelGGL(vtrans_f16_kernel, dim3(BH * (ATT_S / 64)), dim3(256), 0, stream, V, Vt);

  dim3 grid(BH * (ATT_S / 64));   // 1024 workgroups
  dim3 block(128);                // 4 waves, 16 q rows each
  hipLaunchKernelGGL(sdpa_causal_wmma_kernel, grid, block, 0, stream, Qh, Kh, Vt, ctx, attn);
}
